// Attention_20959440404545
// MI455X (gfx1250) — matrix operand order repaired and checked by reading, not compiled, not run on hardware
//
#include <hip/hip_runtime.h>
#include <hip/hip_bf16.h>
#include <math.h>

// ---------------------------------------------------------------------------
// Types for CDNA5 WMMA (wave32): A/B = 16 bf16 per lane, C/D = 8 f32 per lane
// ---------------------------------------------------------------------------
typedef __bf16 v16bf __attribute__((ext_vector_type(16)));
typedef __bf16 v8bf  __attribute__((ext_vector_type(8)));
typedef float  v8f   __attribute__((ext_vector_type(8)));
typedef int    v4i   __attribute__((ext_vector_type(4)));

union ABf { v16bf v; v8bf h[2]; };

#define DIMC  1024
#define NSEQ  2048
#define BATCH 2
#define NH    16
#define CHD   32          // CUR_HD
#define CDIM  512         // CUR_DIM
#define MROWS 4096        // B*N
#define QKVC  1536        // 3*CUR_DIM
#define SCALE_F 0.17677669529663687f   // (64^-0.5)/sqrt(0.5)

#if __has_builtin(__builtin_amdgcn_global_load_async_to_lds_b128)
#define HAVE_ASYNC 1
#endif

__device__ __forceinline__ __bf16 f2bf(float f) {
  unsigned u = __builtin_bit_cast(unsigned, f);
  unsigned r = u + 0x7fffu + ((u >> 16) & 1u);       // round-to-nearest-even
  unsigned short s = (unsigned short)(r >> 16);
  return __builtin_bit_cast(__bf16, s);
}

__device__ __forceinline__ v8f wmma_bf16(v16bf a, v16bf b, v8f c) {
  // (neg_a, A, neg_b, B, c_mod, C, reuse_a, reuse_b)
  return __builtin_amdgcn_wmma_f32_16x16x32_bf16(false, a, false, b,
                                                 (short)0, c, false, false);
}

// 16-byte async global->LDS copy (per-lane addresses), ASYNCcnt-tracked
__device__ __forceinline__ void async_cp16(void* lds, const void* glob) {
#ifdef HAVE_ASYNC
  __builtin_amdgcn_global_load_async_to_lds_b128(
      (__attribute__((address_space(1))) v4i*)(void*)glob,
      (__attribute__((address_space(3))) v4i*)lds, 0, 0);
#else
  *(v8bf*)lds = *(const v8bf*)glob;   // sync fallback: b128 load + ds_store
#endif
}
__device__ __forceinline__ void async_wait1() {
#ifdef HAVE_ASYNC
  asm volatile("s_wait_asynccnt 0x1" ::: "memory");
#endif
}
__device__ __forceinline__ void async_wait0() {
#ifdef HAVE_ASYNC
  asm volatile("s_wait_asynccnt 0x0" ::: "memory");
#endif
}

// ---------------------------------------------------------------------------
// bf16 conversion prep kernels
// ---------------------------------------------------------------------------
__global__ void cvt_f32_bf16(const float* __restrict__ s, __bf16* __restrict__ d, int n) {
  int i = blockIdx.x * blockDim.x + threadIdx.x;
  if (i < n) d[i] = f2bf(s[i]);
}

// w_proj[:, :512] -> row-major (1024 x 512) bf16
__global__ void cvt_wproj(const float* __restrict__ s, __bf16* __restrict__ d) {
  int i = blockIdx.x * blockDim.x + threadIdx.x;
  if (i < DIMC * CDIM) {
    int r = i >> 9, c = i & 511;
    d[i] = f2bf(s[r * DIMC + c]);
  }
}

// ---------------------------------------------------------------------------
// QKV GEMM: Y = X(4096x1024) * Wqkv(1536x1024)^T, scattered into
//   Q[b][h][n][d], K[b][h][n][d], Vt[b][h][d][n]   (bf16)
// One wave computes a 32x32 tile; 4 WMMAs per K=32 step.
// ---------------------------------------------------------------------------
__global__ void __launch_bounds__(256) qkv_gemm_wmma(
    const __bf16* __restrict__ xb, const __bf16* __restrict__ wb,
    __bf16* __restrict__ Qb, __bf16* __restrict__ Kb, __bf16* __restrict__ Vt)
{
  const int lane = threadIdx.x & 31;
  const int wave = (blockIdx.x * 256 + threadIdx.x) >> 5;
  const int tm = wave & 127;          // 128 row tiles
  const int tn = wave >> 7;           // 48 col tiles
  const int row0 = tm * 32, col0 = tn * 32;
  const int half = lane >> 4, l16 = lane & 15;
  const int abase = half ? 16 : 0;
  const int bbase = half ? 16 : 0;

  const __bf16* a0p = xb + (size_t)(row0 + l16) * DIMC;
  const __bf16* a1p = xb + (size_t)(row0 + 16 + l16) * DIMC;
  const __bf16* b0p = wb + (size_t)(col0 + l16) * DIMC;
  const __bf16* b1p = wb + (size_t)(col0 + 16 + l16) * DIMC;

  v8f c00 = {}, c01 = {}, c10 = {}, c11 = {};
  for (int k0 = 0; k0 < DIMC; k0 += 32) {
    if (k0 + 64 < DIMC) {
      __builtin_prefetch(a0p + k0 + 64, 0, 1);
      __builtin_prefetch(a1p + k0 + 64, 0, 1);
    }
    ABf a0, a1;
    a0.h[0] = *(const v8bf*)(a0p + k0 + abase);
    a0.h[1] = *(const v8bf*)(a0p + k0 + 8 + abase);
    a1.h[0] = *(const v8bf*)(a1p + k0 + abase);
    a1.h[1] = *(const v8bf*)(a1p + k0 + 8 + abase);
    v16bf b0 = *(const v16bf*)(b0p + k0 + bbase);
    v16bf b1 = *(const v16bf*)(b1p + k0 + bbase);
    c00 = wmma_bf16(a0.v, b0, c00);
    c01 = wmma_bf16(a0.v, b1, c01);
    c10 = wmma_bf16(a1.v, b0, c10);
    c11 = wmma_bf16(a1.v, b1, c11);
  }

  // 32-wide col block lies entirely within one of {Q,K,V} and one head
  const int which = col0 >> 9;
  const int hc = (col0 & 511) >> 5;
  v8f acc[4] = { c00, c01, c10, c11 };
#pragma unroll
  for (int t = 0; t < 4; ++t) {
    const int roff = (t >> 1) * 16;
    const int d = ((t & 1) * 16) + l16;
#pragma unroll
    for (int j = 0; j < 8; ++j) {
      const int m = row0 + roff + j + half * 8;
      const int b = m >> 11, n = m & 2047;
      const __bf16 val = f2bf(acc[t][j]);
      const size_t hbase = (size_t)b * NH + hc;
      if (which == 0)      Qb[(hbase * NSEQ + n) * CHD + d] = val;
      else if (which == 1) Kb[(hbase * NSEQ + n) * CHD + d] = val;
      else                 Vt[(hbase * CHD + d) * NSEQ + n] = val;
    }
  }
}

// ---------------------------------------------------------------------------
// Flash attention: one wave per 16-row q-block, 8 waves/block sharing (b,h).
// K/V chunks (32 keys) are staged block-cooperatively into LDS with
// double-buffered async global->LDS copies (ASYNCcnt), overlapping the DMA of
// chunk i+1 with the 4 WMMAs + online softmax of chunk i.
// ---------------------------------------------------------------------------
__global__ void __launch_bounds__(256) flash_attn_wmma(
    const __bf16* __restrict__ Qb, const __bf16* __restrict__ Kb,
    const __bf16* __restrict__ Vt, __bf16* __restrict__ Ob)
{
  __shared__ __align__(64) __bf16 kv[2][2][32][32];  // [buf][K=0/V=1][row][32]
  __shared__ __align__(64) __bf16 pbuf[8][16][32];   // per-wave P staging
  const int tid = threadIdx.x;
  const int lane = tid & 31;
  const int wv = tid >> 5;
  const int wave = blockIdx.x * 8 + wv;
  const int bh = wave >> 7;          // same for all 8 waves of a block
  const int qb = wave & 127;
  const int half = lane >> 4, l16 = lane & 15;
  const int abase = half ? 16 : 0;
  const int kbase = half ? 16 : 0;

  // Q tile (A operand), loaded once
  const __bf16* qrow = Qb + ((size_t)bh * NSEQ + qb * 16 + l16) * CHD;
  ABf aq;
  aq.h[0] = *(const v8bf*)(qrow + abase);
  aq.h[1] = *(const v8bf*)(qrow + 8 + abase);

  const __bf16* kbp = Kb + (size_t)bh * NSEQ * CHD;
  const __bf16* vtp = Vt + (size_t)bh * CHD * NSEQ;

  // staging role: threads 0..127 copy the K chunk, 128..255 the V chunk;
  // each thread moves one 16-byte piece per chunk.
  const int cp_isv = tid >> 7;
  const int u = tid & 127;
  const int crow = u >> 2;             // 0..31
  const int cseg = (u & 3) * 8;        // element offset of 16B piece

  auto stage = [&](int p, int kc) {
    if (cp_isv == 0)
      async_cp16(&kv[p][0][crow][cseg], kbp + (size_t)(kc + crow) * CHD + cseg);
    else
      async_cp16(&kv[p][1][crow][cseg], vtp + (size_t)crow * NSEQ + kc + cseg);
  };

  v8f o0 = {}, o1 = {};
  float mrow[8], lrow[8];
#pragma unroll
  for (int j = 0; j < 8; ++j) { mrow[j] = -INFINITY; lrow[j] = 0.f; }

  stage(0, 0);
  int p = 0;
  for (int kc = 0; kc < NSEQ; kc += 32) {
    if (kc + 32 < NSEQ) { stage(p ^ 1, kc + 32); async_wait1(); }
    else                { async_wait0(); }
    __syncthreads();                           // kv[p] visible to all waves

    // S = Q·K^T from LDS: B row = d, col = key
    v16bf bk0 = *(const v16bf*)&kv[p][0][l16][kbase];
    v16bf bk1 = *(const v16bf*)&kv[p][0][16 + l16][kbase];
    v8f s0 = {}, s1 = {};
    s0 = wmma_bf16(aq.v, bk0, s0);   // keys kc..kc+15
    s1 = wmma_bf16(aq.v, bk1, s1);   // keys kc+16..kc+31

    // online softmax; row r = j + 8*half lives on the 16 lanes of this half
#pragma unroll
    for (int j = 0; j < 8; ++j) {
      float e0 = s0[j] * SCALE_F, e1 = s1[j] * SCALE_F;
      float t = fmaxf(e0, e1);
      t = fmaxf(t, __shfl_xor(t, 1, 32));
      t = fmaxf(t, __shfl_xor(t, 2, 32));
      t = fmaxf(t, __shfl_xor(t, 4, 32));
      t = fmaxf(t, __shfl_xor(t, 8, 32));
      float mnew = fmaxf(mrow[j], t);
      float al = __expf(mrow[j] - mnew);
      float p0 = __expf(e0 - mnew);
      float p1 = __expf(e1 - mnew);
      float rs = p0 + p1;
      rs += __shfl_xor(rs, 1, 32);
      rs += __shfl_xor(rs, 2, 32);
      rs += __shfl_xor(rs, 4, 32);
      rs += __shfl_xor(rs, 8, 32);
      lrow[j] = lrow[j] * al + rs;
      mrow[j] = mnew;
      o0[j] *= al;
      o1[j] *= al;
      pbuf[wv][j + half * 8][l16]      = f2bf(p0);
      pbuf[wv][j + half * 8][16 + l16] = f2bf(p1);
    }
    __syncthreads();

    // P re-read in A layout (row = l16, K = local key index)
    ABf pa;
    pa.h[0] = *(const v8bf*)&pbuf[wv][l16][abase];
    pa.h[1] = *(const v8bf*)&pbuf[wv][l16][8 + abase];
    // V^T B tiles from LDS: col = d, K = key row
    v16bf bv0 = *(const v16bf*)&kv[p][1][l16][kbase];
    v16bf bv1 = *(const v16bf*)&kv[p][1][16 + l16][kbase];
    o0 = wmma_bf16(pa.v, bv0, o0);   // d = 0..15
    o1 = wmma_bf16(pa.v, bv1, o1);   // d = 16..31

    // make sure this wave's LDS reads of kv[p] drained before anyone
    // overwrites it (async writes for chunk kc+64 land right after barrier)
    asm volatile("s_wait_dscnt 0x0" ::: "memory");
    __syncthreads();
    p ^= 1;
  }

  const int b = bh >> 4, h = bh & 15;
#pragma unroll
  for (int j = 0; j < 8; ++j) {
    const int r = j + half * 8;
    const int n = qb * 16 + r;
    const float inv = 1.f / lrow[j];
    const size_t base = ((size_t)(b * NSEQ + n)) * CDIM + h * CHD;
    Ob[base + l16]      = f2bf(o0[j] * inv);
    Ob[base + 16 + l16] = f2bf(o1[j] * inv);
  }
}

// ---------------------------------------------------------------------------
// Proj GEMM: out = O(4096x512) * Wp(1024x512)^T + bias, fp32 out
// ---------------------------------------------------------------------------
__global__ void __launch_bounds__(256) proj_gemm_wmma(
    const __bf16* __restrict__ Ob, const __bf16* __restrict__ wpb,
    const float* __restrict__ bias, float* __restrict__ out)
{
  const int lane = threadIdx.x & 31;
  const int wave = (blockIdx.x * 256 + threadIdx.x) >> 5;
  const int tm = wave & 127;
  const int tn = wave >> 7;           // 0..31
  const int row0 = tm * 32, col0 = tn * 32;
  const int half = lane >> 4, l16 = lane & 15;
  const int abase = half ? 16 : 0;
  const int bbase = half ? 16 : 0;

  const __bf16* a0p = Ob + (size_t)(row0 + l16) * CDIM;
  const __bf16* a1p = Ob + (size_t)(row0 + 16 + l16) * CDIM;
  const __bf16* b0p = wpb + (size_t)(col0 + l16) * CDIM;
  const __bf16* b1p = wpb + (size_t)(col0 + 16 + l16) * CDIM;

  v8f c00 = {}, c01 = {}, c10 = {}, c11 = {};
  for (int k0 = 0; k0 < CDIM; k0 += 32) {
    if (k0 + 64 < CDIM) {
      __builtin_prefetch(a0p + k0 + 64, 0, 1);
      __builtin_prefetch(a1p + k0 + 64, 0, 1);
    }
    ABf a0, a1;
    a0.h[0] = *(const v8bf*)(a0p + k0 + abase);
    a0.h[1] = *(const v8bf*)(a0p + k0 + 8 + abase);
    a1.h[0] = *(const v8bf*)(a1p + k0 + abase);
    a1.h[1] = *(const v8bf*)(a1p + k0 + 8 + abase);
    v16bf b0 = *(const v16bf*)(b0p + k0 + bbase);
    v16bf b1 = *(const v16bf*)(b1p + k0 + bbase);
    c00 = wmma_bf16(a0.v, b0, c00);
    c01 = wmma_bf16(a0.v, b1, c01);
    c10 = wmma_bf16(a1.v, b0, c10);
    c11 = wmma_bf16(a1.v, b1, c11);
  }

  const float bv0 = bias[col0 + l16];
  const float bv1 = bias[col0 + 16 + l16];
  v8f acc[4] = { c00, c01, c10, c11 };
#pragma unroll
  for (int t = 0; t < 4; ++t) {
    const int roff = (t >> 1) * 16;
    const int c = col0 + (t & 1) * 16 + l16;
    const float bb = (t & 1) ? bv1 : bv0;
#pragma unroll
    for (int j = 0; j < 8; ++j) {
      const int m = row0 + roff + j + half * 8;
      out[(size_t)m * DIMC + c] = acc[t][j] + bb;
    }
  }
}

// ---------------------------------------------------------------------------
extern "C" void kernel_launch(void* const* d_in, const int* in_sizes, int n_in,
                              void* d_out, int out_size, void* d_ws, size_t ws_size,
                              hipStream_t stream) {
  const float* x      = (const float*)d_in[0];   // (2,2048,1024)
  const float* w_qkv  = (const float*)d_in[1];   // (3072,1024)
  const float* w_proj = (const float*)d_in[2];   // (1024,1024)
  const float* b_proj = (const float*)d_in[3];   // (1024,)
  float* out = (float*)d_out;

  // workspace layout (bf16), ~29.4 MB
  __bf16* xb    = (__bf16*)d_ws;
  __bf16* wqkvb = xb + (size_t)MROWS * DIMC;
  __bf16* wpb   = wqkvb + (size_t)QKVC * DIMC;
  __bf16* Qb    = wpb + (size_t)DIMC * CDIM;
  const size_t hsz = (size_t)BATCH * NH * NSEQ * CHD;   // 2 Mi elems
  __bf16* Kb = Qb + hsz;
  __bf16* Vt = Kb + hsz;
  __bf16* Ob = Vt + hsz;

  hipLaunchKernelGGL(cvt_f32_bf16, dim3((MROWS * DIMC + 255) / 256), dim3(256), 0, stream,
                     x, xb, MROWS * DIMC);
  hipLaunchKernelGGL(cvt_f32_bf16, dim3((QKVC * DIMC + 255) / 256), dim3(256), 0, stream,
                     w_qkv, wqkvb, QKVC * DIMC);
  hipLaunchKernelGGL(cvt_wproj, dim3((DIMC * CDIM + 255) / 256), dim3(256), 0, stream,
                     w_proj, wpb);
  // 128 x 48 tiles of 32x32, 8 waves/block
  hipLaunchKernelGGL(qkv_gemm_wmma, dim3(768), dim3(256), 0, stream, xb, wqkvb, Qb, Kb, Vt);
  // 4096 q-block waves, 8/block (all waves of a block share (b,h))
  hipLaunchKernelGGL(flash_attn_wmma, dim3(512), dim3(256), 0, stream, Qb, Kb, Vt, Ob);
  // 128 x 32 tiles of 32x32
  hipLaunchKernelGGL(proj_gemm_wmma, dim3(512), dim3(256), 0, stream, Ob, wpb, b_proj, out);
}
